// MixedLinearV2_InProj_MHA_75479755260044
// MI455X (gfx1250) — matrix operand order repaired and checked by reading, not compiled, not run on hardware
//
#include <hip/hip_runtime.h>
#include <hip/hip_bf16.h>
#include <stdint.h>

// ---------------------------------------------------------------------------
// MixedLinearV2 in-projection for MI455X (gfx1250, wave32, WMMA bf16)
//
// out[M=8192, N=6144] = X[M,K=1024] @ W_mix^T + b_mix
//   W_mix[o,i] = W[o,i] * sum_k w[k]*(o<out_k)*(i<in_k)   (zero for o>=3072)
//   b_mix[o]   = b[o]   * sum_k w[k]*(o<out_k)
//
// GEMM path: bf16 WMMA, 128x256 block tile (8 waves, wave tile 64x64),
// double-buffered LDS fed by global_load_async_to_lds_b128 (ASYNCcnt).
// ---------------------------------------------------------------------------

typedef __bf16 bf16x16 __attribute__((ext_vector_type(16)));
typedef float  f32x8   __attribute__((ext_vector_type(8)));
typedef unsigned int u32x4 __attribute__((ext_vector_type(4)));

#define M_TOTAL 8192
#define K_TOTAL 1024
#define N_FULL  6144
#define N_GEMM  3072

#define BM 128
#define BN 256
#define BK 32
#define KSTEPS (K_TOTAL / BK)
#define LDSW 40   // bf16 elems per LDS row (32 + 8 pad -> 80B stride, bank stagger)

__device__ __forceinline__ unsigned short f2bf(float f) {
    unsigned int u = __float_as_uint(f);
    unsigned int r = u + 0x7FFFu + ((u >> 16) & 1u);   // round-to-nearest-even
    return (unsigned short)(r >> 16);
}

__device__ __forceinline__ unsigned int pack2(float lo, float hi) {
    return (unsigned int)f2bf(lo) | ((unsigned int)f2bf(hi) << 16);
}

// ---- Kernel 1: X (f32) -> Xb (bf16), 8 elems/thread -----------------------
__global__ __launch_bounds__(256) void k_cvt_x(const float* __restrict__ x,
                                               unsigned short* __restrict__ xb) {
    long idx = ((long)blockIdx.x * 256 + threadIdx.x) * 8;
    const float4* p = (const float4*)(x + idx);
    float4 a = p[0], b = p[1];
    u32x4 o;
    o.x = pack2(a.x, a.y);
    o.y = pack2(a.z, a.w);
    o.z = pack2(b.x, b.y);
    o.w = pack2(b.z, b.w);
    *(u32x4*)(xb + idx) = o;
}

// ---- Kernel 2: W_mix (bf16) for rows [0, N_GEMM) --------------------------
__global__ __launch_bounds__(256) void k_mix_w(const float* __restrict__ W,
                                               const float* __restrict__ wts,
                                               unsigned short* __restrict__ wb) {
    const int OUTK[6] = {1536, 1536, 2304, 2304, 3072, 3072};
    const int INK [6] = { 512,  512,  768,  768, 1024, 1024};
    long idx = ((long)blockIdx.x * 256 + threadIdx.x) * 8;
    int o  = (int)(idx >> 10);          // /1024
    int i0 = (int)(idx & 1023);
    float w[6];
#pragma unroll
    for (int k = 0; k < 6; ++k) w[k] = wts[k];
    const float4* p = (const float4*)(W + (long)o * K_TOTAL + i0);
    float4 a = p[0], b = p[1];
    float v[8] = {a.x, a.y, a.z, a.w, b.x, b.y, b.z, b.w};
    unsigned int packed[4];
#pragma unroll
    for (int e = 0; e < 8; e += 2) {
        float s0 = 0.f, s1 = 0.f;
#pragma unroll
        for (int k = 0; k < 6; ++k) {
            float rk = (o < OUTK[k]) ? w[k] : 0.f;
            s0 += (i0 + e     < INK[k]) ? rk : 0.f;
            s1 += (i0 + e + 1 < INK[k]) ? rk : 0.f;
        }
        packed[e >> 1] = pack2(v[e] * s0, v[e + 1] * s1);
    }
    u32x4 ov = {packed[0], packed[1], packed[2], packed[3]};
    *(u32x4*)(wb + idx) = ov;
}

// ---- Kernel 3: b_mix ------------------------------------------------------
__global__ __launch_bounds__(256) void k_bmix(const float* __restrict__ b,
                                              const float* __restrict__ wts,
                                              float* __restrict__ bmix) {
    const int OUTK[6] = {1536, 1536, 2304, 2304, 3072, 3072};
    int o = blockIdx.x * 256 + threadIdx.x;
    float s = 0.f;
#pragma unroll
    for (int k = 0; k < 6; ++k) s += (o < OUTK[k]) ? wts[k] : 0.f;
    bmix[o] = b[o] * s;
}

// ---- Kernel 4: bf16 WMMA GEMM, async-to-LDS double buffered ---------------
union Frag { bf16x16 v; u32x4 u[2]; };

__global__ __launch_bounds__(256) void k_gemm(const unsigned short* __restrict__ xb,
                                              const unsigned short* __restrict__ wb,
                                              const float* __restrict__ bmix,
                                              float* __restrict__ out) {
    __shared__ unsigned short Abuf[2][BM * LDSW];   // 2 * 10240 B
    __shared__ unsigned short Bbuf[2][BN * LDSW];   // 2 * 20480 B

    const int tid  = threadIdx.x;
    const int lane = tid & 31;
    const int wave = tid >> 5;
    const int half = lane >> 4;      // 0: lanes 0-15, 1: lanes 16-31
    const int lrow = lane & 15;
    const int wm = wave & 1;         // 2 waves along M  -> 64 rows each
    const int wn = wave >> 1;        // 4 waves along N  -> 64 cols each
    const int bm = blockIdx.y * BM;
    const int bn = blockIdx.x * BN;

    const int lr = tid >> 2;   // 0..63: tile row for cooperative copy
    const int lc = tid & 3;    // 16B chunk within 64B (32 bf16) row segment

    // Issue 6 async b128 copies (2 for A, 4 for B) for K-tile k0 into buffer `buf`.
    auto issue_copies = [&](int buf, int k0) {
#pragma unroll
        for (int r = 0; r < 2; ++r) {
            int row = lr + 64 * r;
            unsigned lds = (unsigned)(unsigned long long)(void*)
                           (&Abuf[buf][row * LDSW + lc * 8]);
            unsigned long long ga = (unsigned long long)(void*)
                           (xb + (long)(bm + row) * K_TOTAL + k0 + lc * 8);
            asm volatile("global_load_async_to_lds_b128 %0, %1, off"
                         :: "v"(lds), "v"(ga) : "memory");
        }
#pragma unroll
        for (int r = 0; r < 4; ++r) {
            int row = lr + 64 * r;
            unsigned lds = (unsigned)(unsigned long long)(void*)
                           (&Bbuf[buf][row * LDSW + lc * 8]);
            unsigned long long ga = (unsigned long long)(void*)
                           (wb + (long)(bn + row) * K_TOTAL + k0 + lc * 8);
            asm volatile("global_load_async_to_lds_b128 %0, %1, off"
                         :: "v"(lds), "v"(ga) : "memory");
        }
    };

    // Accumulators pre-seeded with the bias (constant per lane N-column).
    f32x8 acc[4][4];
#pragma unroll
    for (int j = 0; j < 4; ++j) {
        float bias = bmix[bn + 64 * wn + 16 * j + lrow];
#pragma unroll
        for (int i = 0; i < 4; ++i)
#pragma unroll
            for (int r = 0; r < 8; ++r) acc[i][j][r] = bias;
    }

    issue_copies(0, 0);

    for (int t = 0; t < KSTEPS; ++t) {
        const int cur = t & 1;
        if (t + 1 < KSTEPS) {
            issue_copies(cur ^ 1, (t + 1) * BK);
            // warm L2 two tiles ahead
            if (t + 2 < KSTEPS) {
                __builtin_prefetch(xb + (long)(bm + lr) * K_TOTAL + (t + 2) * BK, 0, 0);
                __builtin_prefetch(wb + (long)(bn + lr) * K_TOTAL + (t + 2) * BK, 0, 0);
            }
            // tile t's 6 copies retire first (in-order): wait until <= 6 outstanding
            asm volatile("s_wait_asynccnt 0x6" ::: "memory");
        } else {
            asm volatile("s_wait_asynccnt 0x0" ::: "memory");
        }
        __syncthreads();   // tile t resident for all waves

        const unsigned short* As = Abuf[cur];
        const unsigned short* Bs = Bbuf[cur];

        // B fragments: lane (n = lrow) holds col N; K 0..15 (half=0) / 16..31 (half=1)
        Frag bf[4];
#pragma unroll
        for (int j = 0; j < 4; ++j) {
            const unsigned short* bp = Bs + (64 * wn + 16 * j + lrow) * LDSW + 16 * half;
            bf[j].u[0] = *(const u32x4*)(bp);
            bf[j].u[1] = *(const u32x4*)(bp + 8);
        }
        // A fragments: lane holds row M; K 0..15 (half=0) / 16..31 (half=1)
#pragma unroll
        for (int i = 0; i < 4; ++i) {
            Frag af;
            const unsigned short* ap = As + (64 * wm + 16 * i + lrow) * LDSW + 16 * half;
            af.u[0] = *(const u32x4*)(ap);
            af.u[1] = *(const u32x4*)(ap + 8);
#pragma unroll
            for (int j = 0; j < 4; ++j) {
                acc[i][j] = __builtin_amdgcn_wmma_f32_16x16x32_bf16(
                    /*neg_a=*/false, af.v, /*neg_b=*/false, bf[j].v,
                    /*c_mod=*/(short)0, acc[i][j],
                    /*reuse_a=*/false, /*reuse_b=*/false);
            }
        }
        __syncthreads();   // all reads of buf[cur] done before it is refilled
    }

    // Epilogue: C/D layout -> lane holds (M = r + 8*half, N = lrow) for VGPR r
#pragma unroll
    for (int i = 0; i < 4; ++i)
#pragma unroll
        for (int j = 0; j < 4; ++j) {
            int col  = bn + 64 * wn + 16 * j + lrow;
            int row0 = bm + 64 * wm + 16 * i + 8 * half;
            float* op = out + (long)row0 * N_FULL + col;
#pragma unroll
            for (int r = 0; r < 8; ++r)
                op[(long)r * N_FULL] = acc[i][j][r];
        }
}

// ---- Kernel 5: cols [N_GEMM, N_FULL): out = b_mix (structurally zero) -----
__global__ __launch_bounds__(256) void k_fill(const float* __restrict__ bmix,
                                              float* __restrict__ out) {
    long idx = (long)blockIdx.x * 256 + threadIdx.x;      // one float4 each
    long row = idx / ((N_FULL - N_GEMM) / 4);
    int  c   = (int)(idx % ((N_FULL - N_GEMM) / 4));
    int  col = N_GEMM + c * 4;
    float4 v = {bmix[col], bmix[col + 1], bmix[col + 2], bmix[col + 3]};
    *(float4*)(out + row * N_FULL + col) = v;
}

// ---------------------------------------------------------------------------
extern "C" void kernel_launch(void* const* d_in, const int* in_sizes, int n_in,
                              void* d_out, int out_size, void* d_ws, size_t ws_size,
                              hipStream_t stream) {
    const float* x   = (const float*)d_in[0];   // [8,1024,1024]
    const float* wts = (const float*)d_in[1];   // [6]
    const float* W   = (const float*)d_in[2];   // [6144,1024]
    const float* b   = (const float*)d_in[3];   // [6144]
    float* out = (float*)d_out;                 // [8,1024,6144]

    char* ws = (char*)d_ws;
    unsigned short* xb   = (unsigned short*)ws;                               // 16 MiB
    unsigned short* wb   = (unsigned short*)(ws + (size_t)M_TOTAL * K_TOTAL * 2);
    float*          bmix = (float*)(ws + (size_t)M_TOTAL * K_TOTAL * 2
                                       + (size_t)N_GEMM * K_TOTAL * 2);

    k_cvt_x<<<(M_TOTAL * K_TOTAL) / (256 * 8), 256, 0, stream>>>(x, xb);
    k_mix_w<<<(N_GEMM * K_TOTAL) / (256 * 8), 256, 0, stream>>>(W, wts, wb);
    k_bmix <<<N_FULL / 256, 256, 0, stream>>>(b, wts, bmix);

    dim3 grid(N_GEMM / BN, M_TOTAL / BM);       // 12 x 64 = 768 blocks
    k_gemm<<<grid, 256, 0, stream>>>(xb, wb, bmix, out);

    k_fill<<<(int)(((long)M_TOTAL * (N_FULL - N_GEMM) / 4) / 256), 256, 0, stream>>>(bmix, out);
}
